// GroupedQueryAttention_30434138259533
// MI455X (gfx1250) — matrix operand order repaired and checked by reading, not compiled, not run on hardware
//
#include <hip/hip_runtime.h>
#include <hip/hip_bf16.h>

typedef __bf16 bf16_t;
typedef __attribute__((ext_vector_type(16))) __bf16 v16bf;
typedef __attribute__((ext_vector_type(8)))  __bf16 v8bf;
typedef __attribute__((ext_vector_type(4)))  __bf16 v4bf;
typedef __attribute__((ext_vector_type(8)))  float  v8f;
typedef __attribute__((ext_vector_type(4)))  int    v4i;

#define WMMA_BF16(a, b, c) \
  __builtin_amdgcn_wmma_f32_16x16x32_bf16(false, (a), false, (b), (short)0, (c), false, false)

// ---- async load-to-LDS (gfx1250 ASYNCcnt path), with compile-safe fallback --
#if defined(__has_builtin)
#if __has_builtin(__builtin_amdgcn_global_load_async_to_lds_b128)
#define HAVE_ASYNC_LDS 1
#endif
#endif
#ifndef HAVE_ASYNC_LDS
#define HAVE_ASYNC_LDS 0
#endif

#if HAVE_ASYNC_LDS
// Signature (from compiler diagnostic): first param is generic `int4*`.
#define ASYNC_CP_B128(gp, lp) \
  __builtin_amdgcn_global_load_async_to_lds_b128((v4i*)(void*)(gp), (v4i*)(void*)(lp), 0, 0)
static __device__ __forceinline__ void wait_async0() {
#if __has_builtin(__builtin_amdgcn_s_wait_asynccnt)
  __builtin_amdgcn_s_wait_asynccnt(0);
#else
  asm volatile("s_wait_asynccnt 0x0" ::: "memory");
#endif
}
#endif

// ---- wave32 butterfly reductions via ds_swizzle (immediate XOR patterns) ----
template <int PATTERN>
static __device__ __forceinline__ float swz_f32(float v) {
  return __int_as_float(__builtin_amdgcn_ds_swizzle(__float_as_int(v), PATTERN));
}
template <int PATTERN>
static __device__ __forceinline__ void red_max8(float a[8]) {
#pragma unroll
  for (int r = 0; r < 8; ++r) a[r] = fmaxf(a[r], swz_f32<PATTERN>(a[r]));
}
template <int PATTERN>
static __device__ __forceinline__ void red_add8(float a[8]) {
#pragma unroll
  for (int r = 0; r < 8; ++r) a[r] += swz_f32<PATTERN>(a[r]);
}

// ---------------------------------------------------------------------------
// Fragment loaders (TN layout: both operands K-contiguous)
// ---------------------------------------------------------------------------
static __device__ __forceinline__ v16bf load_a_frag(const bf16_t* base,
                                                    int ld, int row0, int k0, int lane) {
  const int row = row0 + (lane & 15);
  const int kb  = k0 + ((lane >> 4) << 3);
  const bf16_t* p = base + (size_t)row * ld + kb;
  v8bf lo = *reinterpret_cast<const v8bf*>(p);
  v8bf hi = *reinterpret_cast<const v8bf*>(p + 16);
  v16bf r;
#pragma unroll
  for (int e = 0; e < 8; ++e) { r[e] = lo[e]; r[e + 8] = hi[e]; }
  return r;
}

static __device__ __forceinline__ v16bf load_b_frag(const bf16_t* base,
                                                    int ld, int n0, int k0, int lane) {
  const int row = n0 + (lane & 15);
  const int kb  = k0 + ((lane >> 4) << 3);
  const bf16_t* p = base + (size_t)row * ld + kb;
  v8bf lo = *reinterpret_cast<const v8bf*>(p);
  v8bf hi = *reinterpret_cast<const v8bf*>(p + 16);
  v16bf r;
#pragma unroll
  for (int e = 0; e < 8; ++e) { r[e] = lo[e]; r[e + 8] = hi[e]; }
  return r;
}

// ---------------------------------------------------------------------------
// f32 -> bf16 bulk convert
// ---------------------------------------------------------------------------
__global__ void cvt_f32_bf16(const float* __restrict__ in, bf16_t* __restrict__ out, int n4) {
  int i = blockIdx.x * blockDim.x + threadIdx.x;
  if (i >= n4) return;
  float4 v = reinterpret_cast<const float4*>(in)[i];
  v4bf o;
  o[0] = (bf16_t)v.x; o[1] = (bf16_t)v.y; o[2] = (bf16_t)v.z; o[3] = (bf16_t)v.w;
  reinterpret_cast<v4bf*>(out)[i] = o;
}

// ---------------------------------------------------------------------------
// TN GEMM: C[M,N] = A[M,K]*B[N,K]^T (+bias). 8 waves (4M x 2N), wave tile
// 32x64 (2x4 WMMA -> 8 WMMAs per 6 fragment loads), block tile 128x128.
// ---------------------------------------------------------------------------
template <bool BF16_OUT, bool HAS_BIAS>
__global__ __launch_bounds__(256) void gemm_tn(const bf16_t* __restrict__ A,
                                               const bf16_t* __restrict__ B,
                                               bf16_t* __restrict__ Cb,
                                               float* __restrict__ Cf,
                                               const float* __restrict__ bias,
                                               int M, int N, int K) {
  const int lane = threadIdx.x & 31;
  const int wid  = threadIdx.x >> 5;
  const int m0 = blockIdx.x * 128 + (wid >> 1) * 32;
  const int n0 = blockIdx.y * 128 + (wid & 1) * 64;

  v8f acc[2][4] = {};

  for (int k0 = 0; k0 < K; k0 += 32) {
    v16bf a0 = load_a_frag(A, K, m0,      k0, lane);
    v16bf a1 = load_a_frag(A, K, m0 + 16, k0, lane);
    v16bf b0 = load_b_frag(B, K, n0,      k0, lane);
    v16bf b1 = load_b_frag(B, K, n0 + 16, k0, lane);
    v16bf b2 = load_b_frag(B, K, n0 + 32, k0, lane);
    v16bf b3 = load_b_frag(B, K, n0 + 48, k0, lane);
    if (k0 + 32 < K) {  // global_prefetch_b8 hints for next K tile
      __builtin_prefetch(A + (size_t)(m0 + (lane & 15)) * K + k0 + 32, 0, 1);
      __builtin_prefetch(B + (size_t)(n0 + (lane & 15)) * K + k0 + 32, 0, 1);
      __builtin_prefetch(B + (size_t)(n0 + 32 + (lane & 15)) * K + k0 + 32, 0, 1);
    }
    acc[0][0] = WMMA_BF16(a0, b0, acc[0][0]);
    acc[0][1] = WMMA_BF16(a0, b1, acc[0][1]);
    acc[0][2] = WMMA_BF16(a0, b2, acc[0][2]);
    acc[0][3] = WMMA_BF16(a0, b3, acc[0][3]);
    acc[1][0] = WMMA_BF16(a1, b0, acc[1][0]);
    acc[1][1] = WMMA_BF16(a1, b1, acc[1][1]);
    acc[1][2] = WMMA_BF16(a1, b2, acc[1][2]);
    acc[1][3] = WMMA_BF16(a1, b3, acc[1][3]);
  }

  const int cn = lane & 15;
  const int rm = (lane >> 4) << 3;
#pragma unroll
  for (int i = 0; i < 2; ++i) {
#pragma unroll
    for (int j = 0; j < 4; ++j) {
      const int n = n0 + j * 16 + cn;
      const float bv = HAS_BIAS ? bias[n] : 0.0f;
#pragma unroll
      for (int r = 0; r < 8; ++r) {
        const int m = m0 + i * 16 + rm + r;
        const float val = acc[i][j][r] + bv;
        if (BF16_OUT) Cb[(size_t)m * N + n] = (bf16_t)val;
        else          Cf[(size_t)m * N + n] = val;
      }
    }
  }
}

// ---------------------------------------------------------------------------
// V transpose: Vp[B,T,G*64] -> Vt[B,G,64,T]
// ---------------------------------------------------------------------------
__global__ void transpose_v(const bf16_t* __restrict__ Vp, bf16_t* __restrict__ Vt, int T) {
  int idx = blockIdx.x * blockDim.x + threadIdx.x;
  const int t = idx & (T - 1);
  int rest = idx / T;
  const int d = rest & 63;
  rest >>= 6;
  const int g = rest & 7;
  const int b = rest >> 3;
  Vt[idx] = Vp[((size_t)b * T + t) * 512 + g * 64 + d];
}

// ---------------------------------------------------------------------------
// Flash-style GQA attention. Block = 4 waves x 16 query rows of one (b,h).
// K/V tiles staged ONCE per block into LDS (async-to-LDS when available),
// shared by all 4 waves; fragments come from ds_load_b128. Online softmax in
// f32 with ds_swizzle butterfly reductions; P re-layout through per-wave LDS.
// ---------------------------------------------------------------------------
__global__ __launch_bounds__(128) void gqa_attn(const bf16_t* __restrict__ Qp,
                                                const bf16_t* __restrict__ Kp,
                                                const bf16_t* __restrict__ Vt,
                                                bf16_t* __restrict__ AO, int T) {
  constexpr int KSTR = 72;  // 32 keys x (64+8) halves
  constexpr int VSTR = 40;  // 64 d    x (32+8) halves
  constexpr int PSTR = 40;  // 16 q    x (32+8) halves, per wave
  __shared__ __align__(16) bf16_t Ktile[32 * KSTR];
  __shared__ __align__(16) bf16_t Vtile[64 * VSTR];
  __shared__ __align__(16) bf16_t Ptile[4 * 16 * PSTR];

  const int tid  = threadIdx.x;
  const int lane = tid & 31;
  const int wid  = tid >> 5;
  const int b = blockIdx.z, h = blockIdx.y, g = h & 7;  // group = head % 8
  const int t0 = blockIdx.x * 64 + wid * 16;

  const bf16_t* qbase = Qp + (size_t)b * T * 2048 + h * 64;   // rows t, ld 2048
  const bf16_t* kbase = Kp + (size_t)b * T * 512  + g * 64;   // rows key, ld 512
  const bf16_t* vbase = Vt + ((size_t)(b * 8 + g) * 64) * T;  // rows d, ld T
  bf16_t* aobase = AO + (size_t)b * T * 2048 + h * 64;
  bf16_t* pw = &Ptile[wid * 16 * PSTR];

  const v16bf qa0 = load_a_frag(qbase, 2048, t0, 0,  lane);
  const v16bf qa1 = load_a_frag(qbase, 2048, t0, 32, lane);

  v8f o0 = {}, o1 = {}, o2 = {}, o3 = {};
  float mrow[8], lrow[8];
#pragma unroll
  for (int r = 0; r < 8; ++r) { mrow[r] = -3.0e38f; lrow[r] = 0.0f; }
  const float scale = 0.125f;  // 1/sqrt(64)

  for (int key0 = 0; key0 < T; key0 += 32) {
    // ---- stage K (32x64) and V (64x32) tiles for the whole block ----------
#pragma unroll
    for (int c = tid; c < 256; c += 128) {
      const int kr = c >> 3, kc = (c & 7) * 8;   // K tile: 32 rows x 8 chunks
      const int vr = c >> 2, vc = (c & 3) * 8;   // V tile: 64 rows x 4 chunks
      const bf16_t* gk = kbase + (size_t)(key0 + kr) * 512 + kc;
      const bf16_t* gv = vbase + (size_t)vr * T + key0 + vc;
      bf16_t* lk = &Ktile[kr * KSTR + kc];
      bf16_t* lv = &Vtile[vr * VSTR + vc];
#if HAVE_ASYNC_LDS
      ASYNC_CP_B128(gk, lk);
      ASYNC_CP_B128(gv, lv);
#else
      *reinterpret_cast<v8bf*>(lk) = *reinterpret_cast<const v8bf*>(gk);
      *reinterpret_cast<v8bf*>(lv) = *reinterpret_cast<const v8bf*>(gv);
#endif
    }
#if HAVE_ASYNC_LDS
    wait_async0();
#endif
    __syncthreads();

    // ---- scores: S = (Q K^T) * scale, from LDS-resident K -----------------
    v8f s0 = {}, s1 = {};
    {
      v16bf kb0 = load_b_frag(Ktile, KSTR, 0,  0,  lane);
      v16bf kb1 = load_b_frag(Ktile, KSTR, 0,  32, lane);
      s0 = WMMA_BF16(qa0, kb0, s0);
      s0 = WMMA_BF16(qa1, kb1, s0);
      v16bf kb2 = load_b_frag(Ktile, KSTR, 16, 0,  lane);
      v16bf kb3 = load_b_frag(Ktile, KSTR, 16, 32, lane);
      s1 = WMMA_BF16(qa0, kb2, s1);
      s1 = WMMA_BF16(qa1, kb3, s1);
    }

    float mx[8];
#pragma unroll
    for (int r = 0; r < 8; ++r) {
      s0[r] *= scale; s1[r] *= scale;
      mx[r] = fmaxf(s0[r], s1[r]);
    }
    red_max8<0x041F>(mx); red_max8<0x081F>(mx);
    red_max8<0x101F>(mx); red_max8<0x201F>(mx);

    float al[8], rs[8];
#pragma unroll
    for (int r = 0; r < 8; ++r) {
      const float mn = fmaxf(mrow[r], mx[r]);
      al[r] = __expf(mrow[r] - mn);
      mrow[r] = mn;
      const float p0 = __expf(s0[r] - mn);
      const float p1 = __expf(s1[r] - mn);
      rs[r] = p0 + p1;
      const int m = ((lane >> 4) << 3) + r;
      pw[m * PSTR + (lane & 15)]      = (bf16_t)p0;
      pw[m * PSTR + 16 + (lane & 15)] = (bf16_t)p1;
    }
    red_add8<0x041F>(rs); red_add8<0x081F>(rs);
    red_add8<0x101F>(rs); red_add8<0x201F>(rs);
#pragma unroll
    for (int r = 0; r < 8; ++r) lrow[r] = lrow[r] * al[r] + rs[r];

    // ---- P as A-fragment from per-wave LDS (same-wave DS stays ordered) ---
    v16bf pa;
    {
      const bf16_t* lp = pw + (lane & 15) * PSTR + ((lane >> 4) << 3);
      v8bf lo = *reinterpret_cast<const v8bf*>(lp);
      v8bf hi = *reinterpret_cast<const v8bf*>(lp + 16);
#pragma unroll
      for (int e = 0; e < 8; ++e) { pa[e] = lo[e]; pa[e + 8] = hi[e]; }
    }

    // ---- O = O*alpha + P V, from LDS-resident V ---------------------------
    v16bf vb0 = load_b_frag(Vtile, VSTR, 0,  0, lane);
    v16bf vb1 = load_b_frag(Vtile, VSTR, 16, 0, lane);
    v16bf vb2 = load_b_frag(Vtile, VSTR, 32, 0, lane);
    v16bf vb3 = load_b_frag(Vtile, VSTR, 48, 0, lane);
#pragma unroll
    for (int r = 0; r < 8; ++r) {
      o0[r] *= al[r]; o1[r] *= al[r]; o2[r] *= al[r]; o3[r] *= al[r];
    }
    o0 = WMMA_BF16(pa, vb0, o0);
    o1 = WMMA_BF16(pa, vb1, o1);
    o2 = WMMA_BF16(pa, vb2, o2);
    o3 = WMMA_BF16(pa, vb3, o3);

    __syncthreads();  // before next iteration overwrites K/V tiles
  }

  const int cn = lane & 15;
#pragma unroll
  for (int r = 0; r < 8; ++r) {
    const float inv = 1.0f / lrow[r];
    const int t = t0 + ((lane >> 4) << 3) + r;
    bf16_t* o = aobase + (size_t)t * 2048;
    o[cn]      = (bf16_t)(o0[r] * inv);
    o[16 + cn] = (bf16_t)(o1[r] * inv);
    o[32 + cn] = (bf16_t)(o2[r] * inv);
    o[48 + cn] = (bf16_t)(o3[r] * inv);
  }
}

// ---------------------------------------------------------------------------
// Host-side orchestration (all on `stream`, graph-capture safe)
// ---------------------------------------------------------------------------
extern "C" void kernel_launch(void* const* d_in, const int* in_sizes, int n_in,
                              void* d_out, int out_size, void* d_ws, size_t ws_size,
                              hipStream_t stream) {
  (void)in_sizes; (void)n_in; (void)out_size; (void)ws_size;
  const int Bn = 2, T = 2048, C = 2048, G = 8, H = 32, D = 64;
  const size_t nBTC  = (size_t)Bn * T * C;
  const size_t nCC   = (size_t)C * C;
  const size_t nGC   = (size_t)G * D * C;
  const size_t nBTGD = (size_t)Bn * T * G * D;
  const int M = Bn * T;  // 4096

  char* ws = (char*)d_ws;
  size_t off = 0;
  auto alloc = [&](size_t elems) {
    bf16_t* p = (bf16_t*)(ws + off);
    off += ((elems * sizeof(bf16_t) + 255) / 256) * 256;
    return p;
  };
  bf16_t* qb  = alloc(nBTC);
  bf16_t* kb  = alloc(nBTC);
  bf16_t* vb  = alloc(nBTC);
  bf16_t* Wqb = alloc(nCC);
  bf16_t* Wkb = alloc(nGC);
  bf16_t* Wvb = alloc(nGC);
  bf16_t* Wpb = alloc(nCC);
  bf16_t* Qp  = alloc(nBTC);    // [B,T,C]
  bf16_t* Kp  = alloc(nBTGD);   // [B,T,G*64]
  bf16_t* Vp  = alloc(nBTGD);   // [B,T,G*64]
  bf16_t* Vt  = alloc(nBTGD);   // [B,G,64,T]
  bf16_t* AO  = alloc(nBTC);    // [B,T,C]

  auto cvt = [&](const void* in, bf16_t* out, size_t n) {
    const int n4 = (int)(n / 4);
    cvt_f32_bf16<<<(n4 + 255) / 256, 256, 0, stream>>>((const float*)in, out, n4);
  };
  cvt(d_in[0], qb,  nBTC);
  cvt(d_in[1], kb,  nBTC);
  cvt(d_in[2], vb,  nBTC);
  cvt(d_in[3], Wqb, nCC);
  cvt(d_in[4], Wkb, nGC);
  cvt(d_in[5], Wvb, nGC);
  cvt(d_in[6], Wpb, nCC);

  const dim3 blk(256);
  gemm_tn<true, false><<<dim3(M / 128, C / 128), blk, 0, stream>>>(
      qb, Wqb, Qp, nullptr, nullptr, M, C, C);
  gemm_tn<true, false><<<dim3(M / 128, (G * D) / 128), blk, 0, stream>>>(
      kb, Wkb, Kp, nullptr, nullptr, M, G * D, C);
  gemm_tn<true, false><<<dim3(M / 128, (G * D) / 128), blk, 0, stream>>>(
      vb, Wvb, Vp, nullptr, nullptr, M, G * D, C);

  transpose_v<<<(int)(nBTGD / 256), 256, 0, stream>>>(Vp, Vt, T);

  gqa_attn<<<dim3(T / 64, H, Bn), 128, 0, stream>>>(Qp, Kp, Vt, AO, T);

  gemm_tn<false, true><<<dim3(M / 128, C / 128), blk, 0, stream>>>(
      AO, Wpb, nullptr, (float*)d_out, (const float*)d_in[7], M, C, C);
}
